// HydraAttention_34935263985787
// MI455X (gfx1250) — matrix operand order repaired and checked by reading, not compiled, not run on hardware
//
#include <hip/hip_runtime.h>

// ---------------- problem constants ----------------
#define B_   8
#define L_   4096
#define DIM  1024
#define NH   8
#define DH   128
#define DD   16
#define MTOT (B_*L_)        // 32768 tokens
#define NTOT 384            // fused Q|K|V output columns
#define KCH  64             // K-chunk staged per LDS round
#define NCHUNK (DIM/KCH)    // 16
#define MBLK 64             // tokens per block (4 m-tiles of 16)

typedef __attribute__((ext_vector_type(16))) __bf16 v16bf;
typedef __attribute__((ext_vector_type(8)))  float  v8f;
typedef __attribute__((ext_vector_type(4)))  int    v4i;

// ---------------- workspace layout (bytes) ----------------
#define WS_W_OFF    0
#define WS_W_BYTES  (NTOT*DIM*2)
#define WS_BIAS_OFF (WS_W_OFF + WS_W_BYTES)
#define WS_KV_OFF   (WS_BIAS_OFF + 1536)
#define WS_Q_OFF    (WS_KV_OFF + 4096)

// ---------------- async global->LDS probe ----------------
#if defined(__has_builtin)
#  if __has_builtin(__builtin_amdgcn_global_load_async_to_lds_b128)
#    define HAS_ASYNC_LDS 1
#  endif
#endif
#ifndef HAS_ASYNC_LDS
#  define HAS_ASYNC_LDS 0
#endif

typedef __attribute__((address_space(1))) v4i gv4i;   // global int4
typedef __attribute__((address_space(3))) v4i lv4i;   // LDS int4

__device__ __forceinline__ void wait_asynccnt0() {
#if defined(__has_builtin) && __has_builtin(__builtin_amdgcn_s_wait_asynccnt)
  __builtin_amdgcn_s_wait_asynccnt(0);
#else
  asm volatile("s_wait_asynccnt 0" ::: "memory");
#endif
}

// =========================================================
// Kernel A: convert weights to bf16 chunked layout, pack bias, zero kv
// =========================================================
__global__ __launch_bounds__(256) void hydra_prep(
    const float* __restrict__ Wq, const float* __restrict__ bq,
    const float* __restrict__ Wk, const float* __restrict__ bk,
    const float* __restrict__ Wv, const float* __restrict__ bv,
    char* __restrict__ ws)
{
  const int idx = blockIdx.x * 256 + threadIdx.x;   // 0 .. 384*1024-1
  __bf16* wsW = (__bf16*)(ws + WS_W_OFF);
  float*  bias = (float*)(ws + WS_BIAS_OFF);
  float*  kvb  = (float*)(ws + WS_KV_OFF);

  const int n = idx >> 10;        // fused output column 0..383
  const int k = idx & 1023;
  const float* Ws = (n < 128) ? Wq : (n < 256) ? Wk : Wv;
  const float wv = Ws[(size_t)(n & 127) * DIM + k];
  // chunked layout: [k/64][n][k%64]
  wsW[((size_t)(k >> 6) * NTOT + n) * KCH + (k & 63)] = (__bf16)wv;

  if (idx < NTOT)
    bias[idx] = (idx < 128) ? bq[idx] : (idx < 256) ? bk[idx - 128] : bv[idx - 256];
  if (idx < B_ * NH * DD)
    kvb[idx] = 0.0f;
}

// =========================================================
// Kernel B: fused QKV GEMM (bf16 WMMA, f32 acc) + K-norm + kv atomics
// One block = 64 tokens x 384 cols; wave w owns head w's Q/K/V column
// tiles across 4 M-tiles (12 accumulators).
// =========================================================
__global__ __launch_bounds__(256) void hydra_gemm_kv(
    const float* __restrict__ x, char* __restrict__ ws)
{
  char*         wsWb = ws + WS_W_OFF;                // bf16 weight chunks
  const float*  bias = (const float*)(ws + WS_BIAS_OFF);
  float*        kvb  = (float*)(ws + WS_KV_OFF);
  float*        Qbuf = (float*)(ws + WS_Q_OFF);

  __shared__ __bf16 ldsA[MBLK * KCH];   //  8 KB: x tile (64 tokens x 64 k)
  __shared__ __bf16 ldsB[NTOT * KCH];   // 48 KB: weight chunk

  const int tid  = threadIdx.x;
  const int w    = tid >> 5;            // wave id 0..7 == head id
  const int l    = tid & 31;            // lane
  const int half = l >> 4;              // 0: lanes 0-15, 1: lanes 16-31
  const int row  = l & 15;              // N (and A row) index within tile
  const int m0   = blockIdx.x * MBLK;   // first global token row
  const int batch = m0 >> 12;           // m0 / L_ (MBLK divides L_)

  v8f accQ[4] = {}; v8f accK[4] = {}; v8f accV[4] = {};

  for (int c = 0; c < NCHUNK; ++c) {
    // ---- stage B first: 48 KB weight chunk ws -> LDS (async DMA path) ----
    {
      char* src  = wsWb + (size_t)c * NTOT * KCH * 2 + (size_t)tid * 192;
      char* dstb = (char*)ldsB + (size_t)tid * 192;
#if HAS_ASYNC_LDS
      #pragma unroll
      for (int i = 0; i < 12; ++i)
        __builtin_amdgcn_global_load_async_to_lds_b128(
            (gv4i*)(src + i * 16), (lv4i*)(dstb + i * 16), 0, 0);
#else
      #pragma unroll
      for (int i = 0; i < 12; ++i)
        ((uint4*)dstb)[i] = ((const uint4*)src)[i];
#endif
    }
    // ---- stage A (overlaps async copy): x[m0..m0+63, c*64..+64) -> bf16 ----
    {
      const int arow = tid >> 2;              // 0..63
      const int acol = (tid & 3) << 4;        // 0,16,32,48
      const float* xp = x + (size_t)(m0 + arow) * DIM + c * KCH + acol;
      __bf16* dst = &ldsA[arow * KCH + acol];
      #pragma unroll
      for (int i = 0; i < 4; ++i) {
        const float4 v4 = ((const float4*)xp)[i];
        dst[4*i+0] = (__bf16)v4.x; dst[4*i+1] = (__bf16)v4.y;
        dst[4*i+2] = (__bf16)v4.z; dst[4*i+3] = (__bf16)v4.w;
      }
    }
    if (c + 1 < NCHUNK)   // hint next x chunk into cache (global_prefetch_b8)
      __builtin_prefetch(x + (size_t)(m0 + (tid & 63)) * DIM + (c + 1) * KCH, 0, 1);

#if HAS_ASYNC_LDS
    wait_asynccnt0();
#endif
    __syncthreads();

    // ---- two 16x16x32 WMMA K-steps per chunk, 12 wmma each ----
    #pragma unroll
    for (int s = 0; s < 2; ++s) {
      // A fragments (16x32 bf16), one per M-tile; ISA layout:
      // VGPR v, half b : K = s*32 + (v>=4)*16 + half*8 + (v&3)*2 + b
      v16bf af[4];
      #pragma unroll
      for (int mt = 0; mt < 4; ++mt) {
        #pragma unroll
        for (int e = 0; e < 16; ++e) {
          const int vv = e >> 1, bb = e & 1;
          const int K = s * 32 + ((vv & 4) ? 16 : 0) + half * 8 + (vv & 3) * 2 + bb;
          af[mt][e] = ldsA[(mt * 16 + row) * KCH + K];
        }
      }
      // B fragments: lane holds col N=row; K = s*32 + (v>=4)*16 + half*8 + (v&3)*2 + b
      #pragma unroll
      for (int j = 0; j < 3; ++j) {
        const int nt = w + j * 8;           // j=0:Q head w, 1:K head w, 2:V head w
        v16bf bfrag;
        #pragma unroll
        for (int e = 0; e < 16; ++e) {
          const int vv = e >> 1, bb = e & 1;
          const int K = s * 32 + ((vv & 4) ? 16 : 0) + half * 8 + (vv & 3) * 2 + bb;
          bfrag[e] = ldsB[(nt * 16 + row) * KCH + K];
        }
        #pragma unroll
        for (int mt = 0; mt < 4; ++mt) {
          if (j == 0)
            accQ[mt] = __builtin_amdgcn_wmma_f32_16x16x32_bf16(false, af[mt], false, bfrag,
                                                               (short)0, accQ[mt], false, false);
          else if (j == 1)
            accK[mt] = __builtin_amdgcn_wmma_f32_16x16x32_bf16(false, af[mt], false, bfrag,
                                                               (short)0, accK[mt], false, false);
          else
            accV[mt] = __builtin_amdgcn_wmma_f32_16x16x32_bf16(false, af[mt], false, bfrag,
                                                               (short)0, accV[mt], false, false);
        }
      }
    }
    __syncthreads();
  }

  // ---- epilogue: C layout => VGPR r holds (M = r + 8*half, N = row) ----
  const float bq_l = bias[w * 16 + row];
  const float bk_l = bias[128 + w * 16 + row];
  const float bv_l = bias[256 + w * 16 + row];

  float p = 0.0f;
  #pragma unroll
  for (int mt = 0; mt < 4; ++mt) {
    // Q: add bias, spill to ws for the epilogue kernel
    #pragma unroll
    for (int r = 0; r < 8; ++r) {
      const int M = mt * 16 + r + half * 8;
      Qbuf[(size_t)(m0 + M) * DH + w * 16 + row] = accQ[mt][r] + bq_l;
    }
    // K: row-wise L2 norm over 16 cols (cross-lane within half-wave),
    // then accumulate k_norm * V over this M-tile's tokens.
    #pragma unroll
    for (int r = 0; r < 8; ++r) {
      const float kvv = accK[mt][r] + bk_l;
      float ss = kvv * kvv;
      ss += __shfl_xor(ss, 1);
      ss += __shfl_xor(ss, 2);
      ss += __shfl_xor(ss, 4);
      ss += __shfl_xor(ss, 8);         // 16-lane row reduction
      const float kn = kvv * rsqrtf(ss);
      p += kn * (accV[mt][r] + bv_l);
    }
  }
  p += __shfl_xor(p, 16);              // combine halves (M even/odd octets)
  if (half == 0)
    atomicAdd(&kvb[(batch * NH + w) * DD + row], p);
}

// =========================================================
// Kernel C: q-normalize, att = q*kv, LayerNorm(d=16), write out
// one thread per (b, h, t)
// =========================================================
__global__ __launch_bounds__(256) void hydra_epilogue(
    const char* __restrict__ ws, const float* __restrict__ gamma,
    const float* __restrict__ beta, float* __restrict__ out)
{
  const int u  = blockIdx.x * 256 + threadIdx.x;  // < B*NH*L
  const float* Qbuf = (const float*)(ws + WS_Q_OFF);
  const float* kvb  = (const float*)(ws + WS_KV_OFF);
  const int t  = u & (L_ - 1);
  const int bh = u >> 12;          // b*NH + h
  const int h  = bh & (NH - 1);
  const int b  = bh >> 3;

  const float* qp = Qbuf + (size_t)(b * L_ + t) * DH + h * DD;
  float q[DD];
  #pragma unroll
  for (int i = 0; i < 4; ++i) {
    const float4 v4 = ((const float4*)qp)[i];
    q[4*i+0] = v4.x; q[4*i+1] = v4.y; q[4*i+2] = v4.z; q[4*i+3] = v4.w;
  }
  float ss = 0.0f;
  #pragma unroll
  for (int j = 0; j < DD; ++j) ss += q[j] * q[j];
  const float rn = rsqrtf(ss);

  float att[DD]; float mu = 0.0f;
  #pragma unroll
  for (int j = 0; j < DD; ++j) {
    att[j] = q[j] * rn * kvb[bh * DD + j];
    mu += att[j];
  }
  mu *= (1.0f / DD);
  float var = 0.0f;
  #pragma unroll
  for (int j = 0; j < DD; ++j) { const float d = att[j] - mu; var += d * d; }
  var *= (1.0f / DD);
  const float isv = rsqrtf(var + 1e-5f);

  float* op = out + (size_t)bh * (L_ * DD) + (size_t)t * DD;
  #pragma unroll
  for (int j = 0; j < DD; ++j)
    op[j] = (att[j] - mu) * isv * gamma[j] + beta[j];
}

// =========================================================
extern "C" void kernel_launch(void* const* d_in, const int* in_sizes, int n_in,
                              void* d_out, int out_size, void* d_ws, size_t ws_size,
                              hipStream_t stream) {
  const float* x     = (const float*)d_in[0];
  const float* Wq    = (const float*)d_in[1];
  const float* bq    = (const float*)d_in[2];
  const float* Wk    = (const float*)d_in[3];
  const float* bk    = (const float*)d_in[4];
  const float* Wv    = (const float*)d_in[5];
  const float* bv    = (const float*)d_in[6];
  const float* gamma = (const float*)d_in[7];
  const float* beta  = (const float*)d_in[8];
  char* ws  = (char*)d_ws;
  float* out = (float*)d_out;

  hydra_prep<<<(NTOT * DIM) / 256, 256, 0, stream>>>(Wq, bq, Wk, bk, Wv, bv, ws);
  hydra_gemm_kv<<<MTOT / MBLK, 256, 0, stream>>>(x, ws);
  hydra_epilogue<<<(B_ * NH * L_) / 256, 256, 0, stream>>>(ws, gamma, beta, out);
}
